// FastRCNN_42803644072138
// MI455X (gfx1250) — matrix operand order repaired and checked by reading, not compiled, not run on hardware
//
#include <hip/hip_runtime.h>
#include <hip/hip_bf16.h>
#include <cstdint>
#include <cstddef>

// ---------------- CDNA5 vector types ----------------
typedef __attribute__((ext_vector_type(16))) __bf16 v16bf;
typedef __attribute__((ext_vector_type(8)))  __bf16 v8bf;
typedef __attribute__((ext_vector_type(8)))  float  v8f;

// ---------------- problem constants ----------------
#define K_C     512
#define K_HF    50
#define K_WF    84
#define K_POOL  7
#define K_KMAX  5
#define K_NROI  1024
#define K_D     (K_C * K_POOL * K_POOL)   /* 25088 */
#define K_HID   4096
#define K_NEG   (-1e30f)

// GEMM tiling: block 128x128x32, 8 waves (wave32), wave tile 64x32 (4x2 wmma tiles)
#define BM 128
#define BN 128
#define BK 32
#define LDS_STRIDE (BK + 8)   /* 40 halves = 80B row pitch, keeps b128 frags 16B aligned */

// ---------------- fp32 -> bf16 (round to nearest even) ----------------
static __device__ __forceinline__ unsigned short f2bf(float f) {
  union { float f; unsigned int u; } x; x.f = f;
  unsigned int u = x.u;
  unsigned int r = (u + 0x7fffu + ((u >> 16) & 1u)) >> 16;
  return (unsigned short)r;
}

// ---------------- gfx1250 async global->LDS (guarded) ----------------
#if defined(__has_builtin)
#  if __has_builtin(__builtin_amdgcn_global_load_async_to_lds_b128)
#    define CDNA5_ASYNC 1
#  endif
#endif
#ifndef CDNA5_ASYNC
#  define CDNA5_ASYNC 0
#endif

#if CDNA5_ASYNC
// prototype (from clang diagnostic): (v4i AS1*, v4i AS3*, imm offset, imm cpol)
typedef int v4i_gcc __attribute__((vector_size(16)));
typedef __attribute__((address_space(1))) v4i_gcc glb_v4i;
typedef __attribute__((address_space(3))) v4i_gcc lds_v4i;
#endif

static __device__ __forceinline__ void cp_lds_32B(unsigned short* ldst,
                                                  const unsigned short* gsrc) {
#if CDNA5_ASYNC
  glb_v4i* g = (glb_v4i*)(uintptr_t)gsrc;                 // global AS1 (64-bit)
  lds_v4i* l = (lds_v4i*)(unsigned int)(uintptr_t)ldst;   // LDS AS3 (32-bit offset)
  __builtin_amdgcn_global_load_async_to_lds_b128(g, l, 0, 0);
  __builtin_amdgcn_global_load_async_to_lds_b128(g, l, 16, 0);
#else
  ((uint4*)ldst)[0] = ((const uint4*)gsrc)[0];
  ((uint4*)ldst)[1] = ((const uint4*)gsrc)[1];
#endif
}

static __device__ __forceinline__ void stage_wait_and_barrier() {
#if CDNA5_ASYNC
#  if __has_builtin(__builtin_amdgcn_s_wait_asynccnt)
  __builtin_amdgcn_s_wait_asynccnt(0);
#  else
  asm volatile("s_wait_asynccnt 0" ::: "memory");
#  endif
#endif
  __syncthreads();
}

// ---------------- WMMA fragment loads from LDS ----------------
// A (16x32 bf16, M x K): lanes 0-15 -> row M=lane, elems 0..7 = K0..7, 8..15 = K16..23
//                        lanes 16-31 -> row M=lane-16, elems 0..7 = K8..15, 8..15 = K24..31
static __device__ __forceinline__ v16bf load_frag_a(const unsigned short* S, int rbase, int lane) {
  const int r  = rbase + (lane & 15);
  const int k0 = (lane >> 4) * 8;
  const unsigned short* p = S + r * LDS_STRIDE;
  v8bf lo = *(const v8bf*)(p + k0);
  v8bf hi = *(const v8bf*)(p + 16 + k0);
  v16bf o;
#pragma unroll
  for (int i = 0; i < 8; ++i) { o[i] = lo[i]; o[i + 8] = hi[i]; }
  return o;
}

// B (32x16 bf16, K x N) from N-major LDS tile Bt[n][k]:
// lanes 0-15 -> col N=lane, elems 0..7 = K0..7, 8..15 = K16..23 ; lanes 16-31 -> col N=lane-16, elems 0..7 = K8..15, 8..15 = K24..31
static __device__ __forceinline__ v16bf load_frag_b(const unsigned short* S, int rbase, int lane) {
  const int r  = rbase + (lane & 15);
  const int k0 = (lane >> 4) * 8;
  const unsigned short* p = S + r * LDS_STRIDE + k0;
  v8bf lo = *(const v8bf*)(p);
  v8bf hi = *(const v8bf*)(p + 16);
  v16bf o;
#pragma unroll
  for (int i = 0; i < 8; ++i) { o[i] = lo[i]; o[i + 8] = hi[i]; }
  return o;
}

// ---------------- Kernel 1: ROI max-pool -> bf16 pooled[N][25088] ----------------
__global__ __launch_bounds__(256) void roipool_kernel(const float* __restrict__ fm,
                                                      const int* __restrict__ roi,
                                                      unsigned short* __restrict__ out) {
  const int n = blockIdx.x;
  int x0 = roi[n * 4 + 0] / 16, y0 = roi[n * 4 + 1] / 16;
  int x1 = roi[n * 4 + 2] / 16, y1 = roi[n * 4 + 3] / 16;
  if (x1 - x0 <= 1) { if (x1 < K_WF) x1 += 1; else x0 -= 1; }
  if (y1 - y0 <= 1) { if (y1 < K_HF) y1 += 1; else y0 -= 1; }
  const int xs = min(max(x0, 0), K_WF), xe = min(max(x1, 0), K_WF);
  const int ys = min(max(y0, 0), K_HF), ye = min(max(y1, 0), K_HF);
  const int Lx = xe - xs, Ly = ye - ys;

  for (int idx = threadIdx.x; idx < K_D; idx += blockDim.x) {
    const int c = idx / 49, b = idx % 49, i = b / 7, j = b % 7;
    const int hs = ys + i * Ly / K_POOL;
    const int he = ys + ((i + 1) * Ly + K_POOL - 1) / K_POOL;
    const int ws_ = xs + j * Lx / K_POOL;
    const int we = xs + ((j + 1) * Lx + K_POOL - 1) / K_POOL;
    float m = K_NEG;
    const float* f = fm + (size_t)c * K_HF * K_WF;
    for (int kh = 0; kh < K_KMAX; ++kh) {
      const int h = hs + kh;
      if (h >= he) break;
      for (int kw = 0; kw < K_KMAX; ++kw) {
        const int w = ws_ + kw;
        if (w >= we) break;
        m = fmaxf(m, f[h * K_WF + w]);
      }
    }
    out[(size_t)n * K_D + idx] = f2bf(m);
  }
}

// ---------------- Kernel 2: W[K][N] f32 -> Wt[N][K] bf16 (tiled transpose) ----------------
__global__ __launch_bounds__(256) void transpose_cvt_kernel(const float* __restrict__ W,
                                                            unsigned short* __restrict__ Wt,
                                                            int K, int N) {
  __shared__ float t[32][33];
  const int n0 = blockIdx.x * 32, k0 = blockIdx.y * 32;
  const int tx = threadIdx.x, ty = threadIdx.y;
#pragma unroll
  for (int i = 0; i < 32; i += 8)
    t[ty + i][tx] = W[(size_t)(k0 + ty + i) * N + (n0 + tx)];
  __syncthreads();
#pragma unroll
  for (int i = 0; i < 32; i += 8)
    Wt[(size_t)(n0 + ty + i) * K + (k0 + tx)] = f2bf(t[tx][ty + i]);
}

// ---------------- Kernel 3: bf16 WMMA GEMM: C = A[M][K] * Bt[N][K]^T + bias ----------------
__global__ __launch_bounds__(256) void gemm_bf16_kernel(
    const unsigned short* __restrict__ A,   // [M][K] bf16 row-major
    const unsigned short* __restrict__ Bt,  // [N][K] bf16 (weights, pre-transposed)
    const float* __restrict__ bias,         // [N]
    float* __restrict__ Cf,                 // optional f32 out [M][N]
    unsigned short* __restrict__ Cbf,       // optional bf16 out [M][N]
    int M, int N, int K) {
  __shared__ unsigned short As[BM * LDS_STRIDE];
  __shared__ unsigned short Bs[BN * LDS_STRIDE];

  const int tid  = threadIdx.x;
  const int lane = tid & 31;
  const int wave = tid >> 5;       // 8 waves
  const int wm   = wave >> 2;      // 0..1 -> 64-row slab
  const int wn   = wave & 3;       // 0..3 -> 32-col slab
  const int bm   = blockIdx.y * BM;
  const int bn   = blockIdx.x * BN;

  v8f acc[4][2];
  const v8f vzero = {0.f, 0.f, 0.f, 0.f, 0.f, 0.f, 0.f, 0.f};
#pragma unroll
  for (int i = 0; i < 4; ++i)
#pragma unroll
    for (int j = 0; j < 2; ++j) acc[i][j] = vzero;

  // staging: 256 threads, 2 threads per row, 32B each (128 rows x 32 halves)
  const int srow = tid >> 1;
  const int soff = (tid & 1) * 16;
  const unsigned short* gA = A  + (long long)(bm + srow) * K + soff;
  const unsigned short* gB = Bt + (long long)(bn + srow) * K + soff;
  unsigned short* sA = As + srow * LDS_STRIDE + soff;
  unsigned short* sB = Bs + srow * LDS_STRIDE + soff;

  for (int k0 = 0; k0 < K; k0 += BK) {
    cp_lds_32B(sA, gA + k0);
    cp_lds_32B(sB, gB + k0);
    if (k0 + BK < K) {                       // warm L2 for next K tile
      __builtin_prefetch(gA + k0 + BK, 0, 0);
      __builtin_prefetch(gB + k0 + BK, 0, 0);
    }
    stage_wait_and_barrier();

    v16bf af[4], bfr[2];
#pragma unroll
    for (int mt = 0; mt < 4; ++mt) af[mt]  = load_frag_a(As, wm * 64 + mt * 16, lane);
#pragma unroll
    for (int nt = 0; nt < 2; ++nt) bfr[nt] = load_frag_b(Bs, wn * 32 + nt * 16, lane);

#pragma unroll
    for (int mt = 0; mt < 4; ++mt)
#pragma unroll
      for (int nt = 0; nt < 2; ++nt)
        acc[mt][nt] = __builtin_amdgcn_wmma_f32_16x16x32_bf16(
            false, af[mt], false, bfr[nt], (short)0, acc[mt][nt], false, false);

    __syncthreads();   // protect LDS before next stage
  }

  // epilogue: C/D layout -> lane<16: M=r, N=lane ; lane>=16: M=8+r, N=lane-16
  const int nl = lane & 15;
  const int hi = lane >> 4;
#pragma unroll
  for (int mt = 0; mt < 4; ++mt) {
#pragma unroll
    for (int nt = 0; nt < 2; ++nt) {
      const int gn = bn + wn * 32 + nt * 16 + nl;
      const float bv = bias ? bias[gn] : 0.f;
#pragma unroll
      for (int r = 0; r < 8; ++r) {
        const int gm = bm + wm * 64 + mt * 16 + r + 8 * hi;
        const float v = acc[mt][nt][r] + bv;
        if (Cf)  Cf[(size_t)gm * N + gn]  = v;
        if (Cbf) Cbf[(size_t)gm * N + gn] = f2bf(v);
      }
    }
  }
}

// ---------------- Kernel 4: cls/reg heads + softmax ----------------
__global__ __launch_bounds__(128) void heads_kernel(const float* __restrict__ h,
                                                    const float* __restrict__ Wc,
                                                    const float* __restrict__ bc,
                                                    const float* __restrict__ Wr,
                                                    const float* __restrict__ br,
                                                    float* __restrict__ out_cls,
                                                    float* __restrict__ out_reg) {
  const int n = blockIdx.x, tid = threadIdx.x;
  float p[15];
#pragma unroll
  for (int o = 0; o < 15; ++o) p[o] = 0.f;
  const float* hr = h + (size_t)n * K_HID;
  for (int k = tid; k < K_HID; k += 128) {
    const float hv = hr[k];
#pragma unroll
    for (int o = 0; o < 3; ++o)  p[o]     += hv * Wc[k * 3 + o];
#pragma unroll
    for (int o = 0; o < 12; ++o) p[3 + o] += hv * Wr[k * 12 + o];
  }
  __shared__ float red[15][128];
#pragma unroll
  for (int o = 0; o < 15; ++o) red[o][tid] = p[o];
  __syncthreads();
  for (int s = 64; s > 0; s >>= 1) {
    if (tid < s) {
#pragma unroll
      for (int o = 0; o < 15; ++o) red[o][tid] += red[o][tid + s];
    }
    __syncthreads();
  }
  if (tid == 0) {
    const float c0 = red[0][0] + bc[0], c1 = red[1][0] + bc[1], c2 = red[2][0] + bc[2];
    const float mx = fmaxf(c0, fmaxf(c1, c2));
    const float e0 = __expf(c0 - mx), e1 = __expf(c1 - mx), e2 = __expf(c2 - mx);
    const float inv = 1.f / (e0 + e1 + e2);
    out_cls[n * 3 + 0] = e0 * inv;
    out_cls[n * 3 + 1] = e1 * inv;
    out_cls[n * 3 + 2] = e2 * inv;
#pragma unroll
    for (int o = 0; o < 12; ++o) out_reg[n * 12 + o] = red[3 + o][0] + br[o];
  }
}

// ---------------- host launch ----------------
extern "C" void kernel_launch(void* const* d_in, const int* in_sizes, int n_in,
                              void* d_out, int out_size, void* d_ws, size_t ws_size,
                              hipStream_t stream) {
  const float* featmap = (const float*)d_in[0];
  const int*   roi     = (const int*)d_in[1];
  const float* W1 = (const float*)d_in[2];
  const float* b1 = (const float*)d_in[3];
  const float* W2 = (const float*)d_in[4];
  const float* b2 = (const float*)d_in[5];
  const float* Wc = (const float*)d_in[6];
  const float* bc = (const float*)d_in[7];
  const float* Wr = (const float*)d_in[8];
  const float* br = (const float*)d_in[9];

  // workspace layout (bf16 staging + activations), ~316 MB total
  char* ws = (char*)d_ws;
  unsigned short* pooled = (unsigned short*)ws;                                   // 1024 x 25088 bf16
  unsigned short* W1t    = (unsigned short*)(ws + (size_t)K_NROI * K_D * 2);      // 4096 x 25088 bf16
  unsigned short* W2t    = (unsigned short*)((char*)W1t + (size_t)K_D * K_HID * 2);   // 4096 x 4096 bf16
  unsigned short* H1b    = (unsigned short*)((char*)W2t + (size_t)K_HID * K_HID * 2); // 1024 x 4096 bf16
  float*          H2     = (float*)((char*)H1b + (size_t)K_NROI * K_HID * 2);     // 1024 x 4096 f32

  // 1) ROI pool -> bf16
  roipool_kernel<<<K_NROI, 256, 0, stream>>>(featmap, roi, pooled);

  // 2) weights -> transposed bf16 (one-time 0.6 GB pass; GEMM then streams bf16 rows)
  transpose_cvt_kernel<<<dim3(K_HID / 32, K_D / 32), dim3(32, 8), 0, stream>>>(W1, W1t, K_D, K_HID);
  transpose_cvt_kernel<<<dim3(K_HID / 32, K_HID / 32), dim3(32, 8), 0, stream>>>(W2, W2t, K_HID, K_HID);

  // 3) FC1: h = pooled @ W1 + b1  (bf16 out for FC2)
  gemm_bf16_kernel<<<dim3(K_HID / BN, K_NROI / BM), 256, 0, stream>>>(
      pooled, W1t, b1, nullptr, H1b, K_NROI, K_HID, K_D);

  // 4) FC2: h2 = h @ W2 + b2  (f32 out for heads)
  gemm_bf16_kernel<<<dim3(K_HID / BN, K_NROI / BM), 256, 0, stream>>>(
      H1b, W2t, b2, H2, nullptr, K_NROI, K_HID, K_HID);

  // 5) softmax cls + reg
  heads_kernel<<<K_NROI, 128, 0, stream>>>(H2, Wc, bc, Wr, br,
                                           (float*)d_out, (float*)d_out + K_NROI * 3);
}
